// CompactPointModel_37512244363657
// MI455X (gfx1250) — compile-verified, matrix operand order checked by reading, not run on hardware
//
#include <hip/hip_runtime.h>
#include <hip/hip_bf16.h>

// ---------------------------------------------------------------------------
// CompactPointModel forward for MI455X (gfx1250), wave32 + WMMA f16 + TDM.
// ---------------------------------------------------------------------------

typedef __attribute__((ext_vector_type(16))) _Float16 v16h;
typedef __attribute__((ext_vector_type(8)))  _Float16 h8v;
typedef __attribute__((ext_vector_type(8)))  float    v8f;
typedef int i32x4 __attribute__((ext_vector_type(4)));
typedef int i32x8 __attribute__((ext_vector_type(8)));

#define Bc   8
#define Nc   4096
#define Hc   128
#define Kc   16
#define Gc   256
#define Cc   32
#define PDc  16
#define IDc  64
#define FUSEDc 512
#define BNc  (Bc*Nc)

__device__ __forceinline__ h8v ld8(const _Float16* p) {
    return *(const h8v*)p;
}

__device__ __forceinline__ v16h mk16(h8v lo, h8v hi) {
    v16h r;
#pragma unroll
    for (int j = 0; j < 8; ++j) { r[j] = lo[j]; r[j + 8] = hi[j]; }
    return r;
}

// Operand fragment for 16x16x32 f16 WMMA from a row-major f16 buffer.
// lane = {m|n} = L&15 selects the row; h = L>>4 selects the K sub-chunk.
__device__ __forceinline__ v16h frag(const _Float16* base, int ld, int ks, int lane) {
    int mn = lane & 15, h = lane >> 4;
    const _Float16* p = base + (size_t)mn * ld + ks + h * 8;
    return mk16(ld8(p), ld8(p + 16));
}

__device__ __forceinline__ v8f WMMA(v16h a, v16h b, v8f c) {
    return __builtin_amdgcn_wmma_f32_16x16x32_f16(false, a, false, b, (short)0, c,
                                                  false, false);
}

__device__ __forceinline__ void zero8(v8f* acc, int n) {
#pragma unroll 8
    for (int i = 0; i < n; ++i)
#pragma unroll
        for (int r = 0; r < 8; ++r) acc[i][r] = 0.f;
}

// ---------------------------------------------------------------------------
// Weight transpose + f16 convert: w [K][Nout] row-major -> wt [NoutPad][Kpad]
// ---------------------------------------------------------------------------
__global__ void k_tr(const float* __restrict__ w, _Float16* __restrict__ wt,
                     int K, int Nout, int Kpad, int NoutPad) {
    int i = blockIdx.x * 256 + threadIdx.x;
    int total = Kpad * NoutPad;
    if (i >= total) return;
    int n = i / Kpad, k = i % Kpad;
    float v = (n < Nout && k < K) ? w[(size_t)k * Nout + n] : 0.f;
    wt[(size_t)n * Kpad + k] = (_Float16)v;
}

// ---------------------------------------------------------------------------
// Point encoder: [x|n](6) -> relu(lin) -> relu(lin 128x128 via WMMA) -> feat16
// block = 128 threads (4 waves), 64 points per block
// ---------------------------------------------------------------------------
__global__ void k_pe(const float* __restrict__ pts, const float* __restrict__ nrm,
                     const float* __restrict__ w1, const float* __restrict__ b1,
                     const _Float16* __restrict__ w2t, const float* __restrict__ b2,
                     _Float16* __restrict__ feat16) {
    __shared__ __align__(16) _Float16 h1[64 * 128];
    int p0 = blockIdx.x * 64;
    int t  = threadIdx.x;
    int lp = t >> 1;
    int c0 = (t & 1) * 64;
    size_t g = (size_t)(p0 + lp);
    float x[6];
    x[0] = pts[g*3]; x[1] = pts[g*3+1]; x[2] = pts[g*3+2];
    x[3] = nrm[g*3]; x[4] = nrm[g*3+1]; x[5] = nrm[g*3+2];
    for (int c = c0; c < c0 + 64; ++c) {
        float v = b1[c];
#pragma unroll
        for (int k = 0; k < 6; ++k) v += x[k] * w1[k * 128 + c];
        h1[lp * 128 + c] = (_Float16)fmaxf(v, 0.f);
    }
    __syncthreads();

    int wave = t >> 5, lane = t & 31;
    int n = lane & 15, h = lane >> 4;
    const _Float16* A = h1 + wave * 16 * 128;
    v8f acc[8]; zero8(acc, 8);
    for (int ks = 0; ks < 4; ++ks) {
        v16h a = frag(A, 128, ks * 32, lane);
#pragma unroll
        for (int nt = 0; nt < 8; ++nt) {
            v16h bf = frag(w2t + nt * 16 * 128, 128, ks * 32, lane);
            acc[nt] = WMMA(a, bf, acc[nt]);
        }
    }
#pragma unroll
    for (int nt = 0; nt < 8; ++nt)
#pragma unroll
        for (int r = 0; r < 8; ++r) {
            int row = p0 + wave * 16 + r + 8 * h;
            float v = acc[nt][r] + b2[nt * 16 + n];
            feat16[(size_t)row * 128 + nt * 16 + n] = (_Float16)fmaxf(v, 0.f);
        }
}

// ---------------------------------------------------------------------------
// KNN: thread-per-query, LDS candidate tiling, unrolled 16-deep insertion.
// ---------------------------------------------------------------------------
__global__ void k_knn(const float* __restrict__ pts, int* __restrict__ idx) {
    __shared__ float tile[256 * 3];
    int q  = blockIdx.x * 256 + threadIdx.x;
    int b  = q >> 12, ql = q & 4095;
    size_t gq = (size_t)q;
    float qx = pts[gq*3], qy = pts[gq*3+1], qz = pts[gq*3+2];
    float bd[16]; int bi[16];
#pragma unroll
    for (int t = 0; t < 16; ++t) { bd[t] = 3.4e38f; bi[t] = 0; }

    for (int t0 = 0; t0 < Nc; t0 += 256) {
        size_t gc = (size_t)b * Nc + t0 + threadIdx.x;
        tile[threadIdx.x*3+0] = pts[gc*3+0];
        tile[threadIdx.x*3+1] = pts[gc*3+1];
        tile[threadIdx.x*3+2] = pts[gc*3+2];
        if (t0 + 256 < Nc)
            __builtin_prefetch(pts + (gc + 256) * 3, 0, 1);
        __syncthreads();
        for (int j = 0; j < 256; ++j) {
            int c = t0 + j;
            float dx = tile[j*3+0] - qx;
            float dy = tile[j*3+1] - qy;
            float dz = tile[j*3+2] - qz;
            float d = dx*dx + dy*dy + dz*dz;
            if (c != ql && d < bd[15]) {
                bd[15] = d; bi[15] = c;
#pragma unroll
                for (int t = 15; t > 0; --t) {
                    if (bd[t] < bd[t-1]) {
                        float td = bd[t]; bd[t] = bd[t-1]; bd[t-1] = td;
                        int   ti = bi[t]; bi[t] = bi[t-1]; bi[t-1] = ti;
                    }
                }
            }
        }
        __syncthreads();
    }
#pragma unroll
    for (int t = 0; t < 16; ++t) idx[(size_t)q * 16 + t] = bi[t];
}

// ---------------------------------------------------------------------------
// Edge MLP: wave-per-point; K=16 neighbors are the M rows. Layer-2 weights
// (32KB f16) are DMA'd into LDS by the Tensor Data Mover, overlapped with
// layer-1 WMMA compute; tensorcnt wait + block barrier publishes them.
// block = 128 (4 waves = 4 points)
// ---------------------------------------------------------------------------
__global__ void k_edge(const float* __restrict__ pts, const float* __restrict__ nrm,
                       const int* __restrict__ idx, const _Float16* __restrict__ feat16,
                       const _Float16* __restrict__ w1t, const float* __restrict__ b1,
                       const _Float16* __restrict__ w2t, const float* __restrict__ b2,
                       _Float16* __restrict__ loc16) {
    __shared__ __align__(16) _Float16 w2s[128 * 128];     // 32KB TDM destination
    __shared__ __align__(16) _Float16 h1buf[4][16 * 128]; // 16KB
    int wave = threadIdx.x >> 5, lane = threadIdx.x & 31;

    // --- issue TDM: global w2t -> LDS w2s (16384 x 2B elements, 1 row tile) ---
    if (wave == 0) {
        unsigned lds = (unsigned)(unsigned long long)(void*)w2s;
        unsigned long long ga = (unsigned long long)(const void*)w2t;
        i32x4 g0;
        g0[0] = 1;                                   // count=1 (valid descriptor)
        g0[1] = (int)lds;                            // lds_addr
        g0[2] = (int)(ga & 0xffffffffull);           // global_addr[31:0]
        g0[3] = (int)(((ga >> 32) & 0x01ffffffull) | 0x80000000u); // addr[56:32] | type=2
        i32x8 g1;
        g1[0] = 0x00010000;  // workgroup_mask=0, data_size=1 (2 bytes)
        g1[1] = 0x40000000;  // tensor_dim0 = 16384 (low 16 bits at [31:16])
        g1[2] = 0x00010000;  // tensor_dim0 hi = 0, tensor_dim1 = 1
        g1[3] = 0x40000000;  // tensor_dim1 hi = 0, tile_dim0 = 16384
        g1[4] = 0x00000001;  // tile_dim1 = 1, tile_dim2 = 0
        g1[5] = 0x00004000;  // tensor_dim0_stride = 16384 (low 32)
        g1[6] = 0x40000000;  // stride0 hi = 0, tensor_dim1_stride = 16384 (low 16)
        g1[7] = 0x00000000;
        asm volatile("tensor_load_to_lds %0, %1" :: "s"(g0), "s"(g1) : "memory");
    }

    int p = blockIdx.x * 4 + wave;
    int b = p >> 12, q = p & 4095;
    int m = lane & 15, h = lane >> 4;
    int nb = idx[(size_t)p * 16 + m];
    size_t gq = (size_t)b * Nc + q;
    size_t gn = (size_t)b * Nc + nb;
    const _Float16* fq = feat16 + gq * 128;
    const _Float16* fn = feat16 + gn * 128;
    float g0f[6];
#pragma unroll
    for (int j = 0; j < 3; ++j) g0f[j]     = pts[gn*3+j] - pts[gq*3+j];
#pragma unroll
    for (int j = 0; j < 3; ++j) g0f[3 + j] = nrm[gn*3+j] - nrm[gq*3+j];

    // ---- layer 1: edge(262, padded 288) -> 128, A built in registers ----
    v8f acc[8]; zero8(acc, 8);
    for (int ks = 0; ks < 9; ++ks) {
        v16h a;
        if (ks < 4) {                       // ctr = feat[q] (broadcast over m)
            int k0 = ks * 32 + h * 8;
            a = mk16(ld8(fq + k0), ld8(fq + k0 + 16));
        } else if (ks < 8) {                // nbr_f - ctr
            int k0 = (ks - 4) * 32 + h * 8;
            h8v lo = ld8(fn + k0)      - ld8(fq + k0);
            h8v hi = ld8(fn + k0 + 16) - ld8(fq + k0 + 16);
            a = mk16(lo, hi);
        } else {                            // geometry (6) + zero pad
#pragma unroll
            for (int j = 0; j < 16; ++j) a[j] = (_Float16)0.f;
            if (h == 0) {
#pragma unroll
                for (int j = 0; j < 6; ++j) a[j] = (_Float16)g0f[j];
            }
        }
#pragma unroll
        for (int nt = 0; nt < 8; ++nt) {
            v16h bf = frag(w1t + nt * 16 * 288, 288, ks * 32, lane);
            acc[nt] = WMMA(a, bf, acc[nt]);
        }
    }
    _Float16* h1 = &h1buf[wave][0];
#pragma unroll
    for (int nt = 0; nt < 8; ++nt)
#pragma unroll
        for (int r = 0; r < 8; ++r) {
            float v = acc[nt][r] + b1[nt * 16 + m];
            h1[(r + 8 * h) * 128 + nt * 16 + m] = (_Float16)fmaxf(v, 0.f);
        }

    // ---- wait for TDM weights, publish to block ----
    if (wave == 0) __builtin_amdgcn_s_wait_tensorcnt(0);
    __syncthreads();

    // ---- layer 2 (K=128), B fragments from LDS ----
    v8f acc2[8]; zero8(acc2, 8);
    for (int ks = 0; ks < 4; ++ks) {
        v16h a = frag(h1, 128, ks * 32, lane);
#pragma unroll
        for (int nt = 0; nt < 8; ++nt) {
            v16h bf = frag(w2s + nt * 16 * 128, 128, ks * 32, lane);
            acc2[nt] = WMMA(a, bf, acc2[nt]);
        }
    }
    // relu + max over the 16 neighbor rows -> loc[q]
#pragma unroll
    for (int nt = 0; nt < 8; ++nt) {
        float cm = -1e30f;
#pragma unroll
        for (int r = 0; r < 8; ++r) {
            float v = fmaxf(acc2[nt][r] + b2[nt * 16 + m], 0.f);
            cm = fmaxf(cm, v);
        }
        cm = fmaxf(cm, __shfl_xor(cm, 16, 32));
        if (h == 0) loc16[gq * 128 + nt * 16 + m] = (_Float16)cm;
    }
}

// ---------------------------------------------------------------------------
// Global pooling: gmax/gmean over N, then relu(gcat @ gp_w + gp_b) -> gfeat16
// ---------------------------------------------------------------------------
__global__ void k_pool(const _Float16* __restrict__ loc16,
                       const float* __restrict__ gpw, const float* __restrict__ gpb,
                       _Float16* __restrict__ gfeat16) {
    __shared__ float gcat[256];
    int b = blockIdx.x, c = threadIdx.x;
    float mx = -1e30f, sm = 0.f;
    for (int n = 0; n < Nc; ++n) {
        float v = (float)loc16[((size_t)b * Nc + n) * 128 + c];
        mx = fmaxf(mx, v); sm += v;
    }
    gcat[c]       = mx;
    gcat[128 + c] = sm * (1.f / (float)Nc);
    __syncthreads();
    for (int o = 0; o < 2; ++o) {
        int n = c + o * 128;
        float v = gpb[n];
        for (int k = 0; k < 256; ++k) v += gcat[k] * gpw[k * 256 + n];
        gfeat16[(size_t)b * 256 + n] = (_Float16)fmaxf(v, 0.f);
    }
}

// fused = [feat | loc | gfeat(broadcast)] in f16
__global__ void k_fused(const _Float16* __restrict__ feat16,
                        const _Float16* __restrict__ loc16,
                        const _Float16* __restrict__ gfeat16,
                        _Float16* __restrict__ fused16) {
    size_t i = (size_t)blockIdx.x * 256 + threadIdx.x;
    int pt = (int)(i >> 9), c = (int)(i & 511);
    _Float16 v;
    if (c < 128)      v = feat16[(size_t)pt * 128 + c];
    else if (c < 256) v = loc16[(size_t)pt * 128 + (c - 128)];
    else              v = gfeat16[(size_t)(pt >> 12) * 256 + (c - 256)];
    fused16[i] = v;
}

// ---------------------------------------------------------------------------
// Head helpers, 2 M-tiles (32 points) per wave: every B fragment is reused
// twice, halving the load:wmma ratio of the dominant 512->128 GEMMs.
// ---------------------------------------------------------------------------
__device__ __forceinline__ void head_hidden2(const _Float16* fusedRow,
                                             const _Float16* w1t, const float* b1,
                                             _Float16* h1, int lane) {
    v8f acc0[8], acc1[8];
    zero8(acc0, 8); zero8(acc1, 8);
    for (int ks = 0; ks < 16; ++ks) {
        v16h a0 = frag(fusedRow,            512, ks * 32, lane);
        v16h a1 = frag(fusedRow + 16 * 512, 512, ks * 32, lane);
#pragma unroll
        for (int nt = 0; nt < 8; ++nt) {
            v16h bf = frag(w1t + nt * 16 * 512, 512, ks * 32, lane);
            acc0[nt] = WMMA(a0, bf, acc0[nt]);
            acc1[nt] = WMMA(a1, bf, acc1[nt]);
        }
    }
    int n = lane & 15, h = lane >> 4;
#pragma unroll
    for (int nt = 0; nt < 8; ++nt)
#pragma unroll
        for (int r = 0; r < 8; ++r) {
            float v0 = acc0[nt][r] + b1[nt * 16 + n];
            float v1 = acc1[nt][r] + b1[nt * 16 + n];
            h1[(r + 8 * h) * 128 + nt * 16 + n]      = (_Float16)fmaxf(v0, 0.f);
            h1[(16 + r + 8 * h) * 128 + nt * 16 + n] = (_Float16)fmaxf(v1, 0.f);
        }
}

// acc2 layout: [mt*NT + nt]
template <int NT>
__device__ __forceinline__ void head_out2(const _Float16* h1, const _Float16* w2t,
                                          v8f* acc2, int lane) {
    zero8(acc2, 2 * NT);
    for (int ks = 0; ks < 4; ++ks) {
        v16h a0 = frag(h1,            128, ks * 32, lane);
        v16h a1 = frag(h1 + 16 * 128, 128, ks * 32, lane);
#pragma unroll
        for (int nt = 0; nt < NT; ++nt) {
            v16h bf = frag(w2t + nt * 16 * 128, 128, ks * 32, lane);
            acc2[nt]      = WMMA(a0, bf, acc2[nt]);
            acc2[NT + nt] = WMMA(a1, bf, acc2[NT + nt]);
        }
    }
}

// par head: 512->128->16   (block 128 = 4 waves = 128 points)
__global__ void k_par(const _Float16* __restrict__ fused16,
                      const _Float16* __restrict__ w1t, const float* __restrict__ b1,
                      const _Float16* __restrict__ w2t, const float* __restrict__ b2,
                      float* __restrict__ out) {
    __shared__ __align__(16) _Float16 h1buf[4][32 * 128];
    int wave = threadIdx.x >> 5, lane = threadIdx.x & 31;
    int pt0 = blockIdx.x * 128 + wave * 32;
    head_hidden2(fused16 + (size_t)pt0 * 512, w1t, b1, &h1buf[wave][0], lane);
    v8f acc2[2];
    head_out2<1>(&h1buf[wave][0], w2t, acc2, lane);
    int n = lane & 15, h = lane >> 4;
#pragma unroll
    for (int mt = 0; mt < 2; ++mt)
#pragma unroll
        for (int r = 0; r < 8; ++r)
            out[(size_t)(pt0 + mt * 16 + r + 8 * h) * 16 + n] = acc2[mt][r] + b2[n];
}

// bnd head: 512->128->1 (w2t padded to 16 rows; only col 0 valid)
__global__ void k_bnd(const _Float16* __restrict__ fused16,
                      const _Float16* __restrict__ w1t, const float* __restrict__ b1,
                      const _Float16* __restrict__ w2t, const float* __restrict__ b2,
                      float* __restrict__ out) {
    __shared__ __align__(16) _Float16 h1buf[4][32 * 128];
    int wave = threadIdx.x >> 5, lane = threadIdx.x & 31;
    int pt0 = blockIdx.x * 128 + wave * 32;
    head_hidden2(fused16 + (size_t)pt0 * 512, w1t, b1, &h1buf[wave][0], lane);
    v8f acc2[2];
    head_out2<1>(&h1buf[wave][0], w2t, acc2, lane);
    int n = lane & 15, h = lane >> 4;
    if (n == 0) {
#pragma unroll
        for (int mt = 0; mt < 2; ++mt)
#pragma unroll
            for (int r = 0; r < 8; ++r)
                out[(size_t)(pt0 + mt * 16 + r + 8 * h)] = acc2[mt][r] + b2[0];
    }
}

// cls head + softmax + ctx = sm @ icp_w; fused += SCALE*ctx (in place)
__global__ void k_cls(_Float16* __restrict__ fused16,
                      const _Float16* __restrict__ w1t, const float* __restrict__ b1,
                      const _Float16* __restrict__ w2t, const float* __restrict__ b2,
                      const _Float16* __restrict__ icpwt, const float* __restrict__ icpb,
                      float* __restrict__ logits_out) {
    __shared__ __align__(16) _Float16 h1buf[4][32 * 128];   // 32KB
    __shared__ float lg[4][32 * 32];                        // 16KB
    __shared__ __align__(16) _Float16 sm16[4][32 * 32];     // 8KB
    int wave = threadIdx.x >> 5, lane = threadIdx.x & 31;
    int pt0 = blockIdx.x * 128 + wave * 32;
    head_hidden2(fused16 + (size_t)pt0 * 512, w1t, b1, &h1buf[wave][0], lane);
    v8f acc2[4];
    head_out2<2>(&h1buf[wave][0], w2t, acc2, lane);
    int n = lane & 15, h = lane >> 4;
#pragma unroll
    for (int mt = 0; mt < 2; ++mt)
#pragma unroll
        for (int nt = 0; nt < 2; ++nt)
#pragma unroll
            for (int r = 0; r < 8; ++r) {
                int row = mt * 16 + r + 8 * h, col = nt * 16 + n;
                float v = acc2[mt * 2 + nt][r] + b2[col];
                logits_out[(size_t)(pt0 + row) * 32 + col] = v;
                lg[wave][row * 32 + col] = v;
            }
    __syncthreads();
    {   // softmax: 32 lanes handle the 32 rows of this wave's tile
        const float* rowp = &lg[wave][lane * 32];
        float mx = -1e30f;
        for (int c = 0; c < 32; ++c) mx = fmaxf(mx, rowp[c]);
        float s = 0.f;
        for (int c = 0; c < 32; ++c) s += __expf(rowp[c] - mx);
        float inv = 1.f / s;
        for (int c = 0; c < 32; ++c)
            sm16[wave][lane * 32 + c] = (_Float16)(__expf(rowp[c] - mx) * inv);
    }
    __syncthreads();
    // ctx: M=2x16 points, K=32 (one step), N=512 (32 n-tiles, B reused twice)
    v16h a0 = frag(&sm16[wave][0],       32, 0, lane);
    v16h a1 = frag(&sm16[wave][16 * 32], 32, 0, lane);
    for (int nt = 0; nt < 32; ++nt) {
        v16h bf = frag(icpwt + nt * 16 * 32, 32, 0, lane);
        v8f z; zero8(&z, 1);
        v8f cc0 = WMMA(a0, bf, z);
        v8f cc1 = WMMA(a1, bf, z);
#pragma unroll
        for (int r = 0; r < 8; ++r) {
            int col = nt * 16 + n;
            size_t o0 = (size_t)(pt0 + r + 8 * h) * 512 + col;
            size_t o1 = (size_t)(pt0 + 16 + r + 8 * h) * 512 + col;
            fused16[o0] = (_Float16)((float)fused16[o0] + 0.1f * (cc0[r] + icpb[col]));
            fused16[o1] = (_Float16)((float)fused16[o1] + 0.1f * (cc1[r] + icpb[col]));
        }
    }
}

// ins head: (fused + SCALE*ctx) -> 128 -> 64, then L2-normalize per point
__global__ void k_ins(const _Float16* __restrict__ fused16,
                      const _Float16* __restrict__ w1t, const float* __restrict__ b1,
                      const _Float16* __restrict__ w2t, const float* __restrict__ b2,
                      float* __restrict__ out) {
    __shared__ __align__(16) _Float16 h1buf[4][32 * 128];   // 32KB, reused below
    int wave = threadIdx.x >> 5, lane = threadIdx.x & 31;
    int pt0 = blockIdx.x * 128 + wave * 32;
    head_hidden2(fused16 + (size_t)pt0 * 512, w1t, b1, &h1buf[wave][0], lane);
    v8f acc2[8];
    head_out2<4>(&h1buf[wave][0], w2t, acc2, lane);
    __syncthreads();                       // h1 reads done; reuse the tile as f32
    float* ib = (float*)&h1buf[wave][0];   // 32 rows x 64 cols f32 == 8KB
    int n = lane & 15, h = lane >> 4;
#pragma unroll
    for (int mt = 0; mt < 2; ++mt)
#pragma unroll
        for (int nt = 0; nt < 4; ++nt)
#pragma unroll
            for (int r = 0; r < 8; ++r)
                ib[(mt * 16 + r + 8 * h) * 64 + nt * 16 + n] =
                    acc2[mt * 4 + nt][r] + b2[nt * 16 + n];
    __syncthreads();
    {   // 32 lanes normalize the 32 rows of this wave's tile
        const float* rowp = &ib[lane * 64];
        float s = 0.f;
        for (int c = 0; c < 64; ++c) s += rowp[c] * rowp[c];
        float inv = 1.f / fmaxf(sqrtf(s), 1e-12f);
        for (int c = 0; c < 64; ++c)
            out[(size_t)(pt0 + lane) * 64 + c] = rowp[c] * inv;
    }
}

// ---------------------------------------------------------------------------
extern "C" void kernel_launch(void* const* d_in, const int* in_sizes, int n_in,
                              void* d_out, int out_size, void* d_ws, size_t ws_size,
                              hipStream_t stream) {
    (void)in_sizes; (void)n_in; (void)out_size; (void)ws_size;
    const float* points  = (const float*)d_in[0];
    const float* normals = (const float*)d_in[1];
    const float* pe_w1 = (const float*)d_in[2];
    const float* pe_b1 = (const float*)d_in[3];
    const float* pe_w2 = (const float*)d_in[4];
    const float* pe_b2 = (const float*)d_in[5];
    const float* lm_w1 = (const float*)d_in[6];
    const float* lm_b1 = (const float*)d_in[7];
    const float* lm_w2 = (const float*)d_in[8];
    const float* lm_b2 = (const float*)d_in[9];
    const float* gp_w  = (const float*)d_in[10];
    const float* gp_b  = (const float*)d_in[11];
    const float* cls_w1 = (const float*)d_in[12];
    const float* cls_b1 = (const float*)d_in[13];
    const float* cls_w2 = (const float*)d_in[14];
    const float* cls_b2 = (const float*)d_in[15];
    const float* par_w1 = (const float*)d_in[16];
    const float* par_b1 = (const float*)d_in[17];
    const float* par_w2 = (const float*)d_in[18];
    const float* par_b2 = (const float*)d_in[19];
    const float* bnd_w1 = (const float*)d_in[20];
    const float* bnd_b1 = (const float*)d_in[21];
    const float* bnd_w2 = (const float*)d_in[22];
    const float* bnd_b2 = (const float*)d_in[23];
    const float* icp_w  = (const float*)d_in[24];
    const float* icp_b  = (const float*)d_in[25];
    const float* ins_w1 = (const float*)d_in[26];
    const float* ins_b1 = (const float*)d_in[27];
    const float* ins_w2 = (const float*)d_in[28];
    const float* ins_b2 = (const float*)d_in[29];

    // workspace carve-up (256B-aligned regions)
    char* ws = (char*)d_ws;
    size_t off = 0;
    auto take = [&](size_t bytes) -> char* {
        char* r = ws + off;
        off = (off + bytes + 255) & ~(size_t)255;
        return r;
    };
    _Float16* feat16  = (_Float16*)take((size_t)BNc * Hc * 2);
    _Float16* loc16   = (_Float16*)take((size_t)BNc * Hc * 2);
    _Float16* fused16 = (_Float16*)take((size_t)BNc * FUSEDc * 2);
    int*      idx     = (int*)take((size_t)BNc * Kc * 4);
    _Float16* gfeat16 = (_Float16*)take((size_t)Bc * Gc * 2);
    _Float16* pe_w2t  = (_Float16*)take(128 * 128 * 2);
    _Float16* lm_w1t  = (_Float16*)take(128 * 288 * 2);
    _Float16* lm_w2t  = (_Float16*)take(128 * 128 * 2);
    _Float16* cls_w1t = (_Float16*)take(128 * 512 * 2);
    _Float16* par_w1t = (_Float16*)take(128 * 512 * 2);
    _Float16* bnd_w1t = (_Float16*)take(128 * 512 * 2);
    _Float16* ins_w1t = (_Float16*)take(128 * 512 * 2);
    _Float16* cls_w2t = (_Float16*)take(32 * 128 * 2);
    _Float16* par_w2t = (_Float16*)take(16 * 128 * 2);
    _Float16* bnd_w2t = (_Float16*)take(16 * 128 * 2);
    _Float16* ins_w2t = (_Float16*)take(64 * 128 * 2);
    _Float16* icp_wt  = (_Float16*)take(512 * 32 * 2);

    auto TR = [&](const float* w, _Float16* wt, int K, int Nout, int Kpad, int NP) {
        int total = Kpad * NP;
        k_tr<<<(total + 255) / 256, 256, 0, stream>>>(w, wt, K, Nout, Kpad, NP);
    };
    TR(pe_w2,  pe_w2t,  128, 128, 128, 128);
    TR(lm_w1,  lm_w1t,  262, 128, 288, 128);
    TR(lm_w2,  lm_w2t,  128, 128, 128, 128);
    TR(cls_w1, cls_w1t, 512, 128, 512, 128);
    TR(par_w1, par_w1t, 512, 128, 512, 128);
    TR(bnd_w1, bnd_w1t, 512, 128, 512, 128);
    TR(ins_w1, ins_w1t, 512, 128, 512, 128);
    TR(cls_w2, cls_w2t, 128, 32, 128, 32);
    TR(par_w2, par_w2t, 128, 16, 128, 16);
    TR(bnd_w2, bnd_w2t, 128, 1, 128, 16);
    TR(ins_w2, ins_w2t, 128, 64, 128, 64);
    TR(icp_w,  icp_wt,  32, 512, 32, 512);

    k_pe  <<<BNc / 64, 128, 0, stream>>>(points, normals, pe_w1, pe_b1, pe_w2t, pe_b2, feat16);
    k_knn <<<BNc / 256, 256, 0, stream>>>(points, idx);
    k_edge<<<BNc / 4, 128, 0, stream>>>(points, normals, idx, feat16,
                                        lm_w1t, lm_b1, lm_w2t, lm_b2, loc16);
    k_pool<<<Bc, 128, 0, stream>>>(loc16, gp_w, gp_b, gfeat16);
    k_fused<<<(BNc * FUSEDc) / 256, 256, 0, stream>>>(feat16, loc16, gfeat16, fused16);

    float* out = (float*)d_out;
    float* logits = out;
    float* prm = out + (size_t)BNc * Cc;
    float* bnd = out + (size_t)BNc * (Cc + PDc);
    float* ins = out + (size_t)BNc * (Cc + PDc + 1);

    // par/bnd read pre-update fused; cls then updates fused in place; ins last.
    k_par<<<BNc / 128, 128, 0, stream>>>(fused16, par_w1t, par_b1, par_w2t, par_b2, prm);
    k_bnd<<<BNc / 128, 128, 0, stream>>>(fused16, bnd_w1t, bnd_b1, bnd_w2t, bnd_b2, bnd);
    k_cls<<<BNc / 128, 128, 0, stream>>>(fused16, cls_w1t, cls_b1, cls_w2t, cls_b2,
                                         icp_wt, icp_b, logits);
    k_ins<<<BNc / 128, 128, 0, stream>>>(fused16, ins_w1t, ins_b1, ins_w2t, ins_b2, ins);
}
